// MILoss_26680336843005
// MI455X (gfx1250) — matrix operand order repaired and checked by reading, not compiled, not run on hardware
//
#include <hip/hip_runtime.h>
#include <hip/hip_bf16.h>

typedef __attribute__((ext_vector_type(16))) __bf16 v16bf;
typedef __attribute__((ext_vector_type(8)))  float  v8f;

#define BD     128      // B (graphs / ctx vectors)
#define SD     1024     // S (tokens per graph)
#define DD     1024     // D (embedding dim)
#define MTILES 8192     // (B*S)/16
#define NT     8        // N tiles (128/16)
#define KSTEPS 32       // 1024/32
#define LOG2F_ 0.6931471805599453f

// ---------------- Kernel 0: pack ctx f32 -> bf16 in B-fragment order ----------
// Layout: [nt(8)][kstep(32)][lane(32)][16 bf16]  -> each lane's 32B is contiguous.
// Lane l (col = l&15, khalf = l>>4) holds K = kstep*32 + khalf*16 + j.
__global__ __launch_bounds__(32) void pack_ctx_kernel(const float* __restrict__ ctx,
                                                      __bf16* __restrict__ ctxb) {
    const int lane  = threadIdx.x;
    const int nt    = blockIdx.x >> 5;
    const int ks    = blockIdx.x & 31;
    const int col   = lane & 15;
    const int khalf = lane >> 4;
    const float* src = ctx + (size_t)(nt * 16 + col) * DD + ks * 32 + khalf * 16;
    const float4 f0 = *(const float4*)(src);
    const float4 f1 = *(const float4*)(src + 4);
    const float4 f2 = *(const float4*)(src + 8);
    const float4 f3 = *(const float4*)(src + 12);
    v16bf v;
    v[0]  = (__bf16)f0.x; v[1]  = (__bf16)f0.y; v[2]  = (__bf16)f0.z; v[3]  = (__bf16)f0.w;
    v[4]  = (__bf16)f1.x; v[5]  = (__bf16)f1.y; v[6]  = (__bf16)f1.z; v[7]  = (__bf16)f1.w;
    v[8]  = (__bf16)f2.x; v[9]  = (__bf16)f2.y; v[10] = (__bf16)f2.z; v[11] = (__bf16)f2.w;
    v[12] = (__bf16)f3.x; v[13] = (__bf16)f3.y; v[14] = (__bf16)f3.z; v[15] = (__bf16)f3.w;
    *(v16bf*)(ctxb + ((size_t)blockIdx.x * 32 + lane) * 16) = v;
}

// ---------------- Kernel 1: per-graph valid lengths ----------------
__global__ __launch_bounds__(256) void lens_kernel(const int* __restrict__ mask,
                                                   int* __restrict__ lens) {
    __shared__ int sdata[256];
    const int b = blockIdx.x;
    const int t = threadIdx.x;
    int sum = 0;
    for (int s = t; s < SD; s += 256) sum += mask[b * SD + s];
    sdata[t] = sum;
    __syncthreads();
    for (int off = 128; off > 0; off >>= 1) {
        if (t < off) sdata[t] += sdata[t + off];
        __syncthreads();
    }
    if (t == 0) lens[b] = sdata[0] < 1 ? 1 : sdata[0];
}

// ---------------- Kernel 2: WMMA GEMM + JSD epilogue ----------------
// grid.x = MTILES/8 ; block = 256 (8 waves). Wave w owns M-tile blk*8+w and
// iterates all 8 N-tiles, so one A fragment feeds 8 WMMAs per K-step.
// Per K-step: batch-load the 8 B fragments (constant immediate offsets),
// prefetch next K-step's A, convert A, then run the 8-WMMA burst.
__global__ __launch_bounds__(256) void miloss_main(const float* __restrict__ seq,
                                                   const __bf16* __restrict__ ctxb,
                                                   const int*   __restrict__ lens,
                                                   float* __restrict__ partials) {
    const int wave  = threadIdx.x >> 5;
    const int lane  = threadIdx.x & 31;
    const int rc    = lane & 15;          // A row within tile
    const int khalf = lane >> 4;

    const int mTile = blockIdx.x * 8 + wave;
    const int tok0  = mTile * 16;         // 16 consecutive tokens, same graph b
    const int b_idx = tok0 >> 10;         // tok / S
    const int s_base = tok0 & (SD - 1);
    const int len = lens[b_idx];

    // A fragment source (f32): lane holds K run [khalf*16,+16)
    const float* pA = seq + (size_t)(tok0 + rc) * DD + khalf * 16;
    // B fragments (pre-packed bf16): fragment (nt, ks) at pB + (nt*KSTEPS+ks)*32
    const v16bf* pBks = (const v16bf*)(ctxb + (size_t)lane * 16);

    v8f acc[NT];
#pragma unroll
    for (int nt = 0; nt < NT; ++nt) acc[nt] = (v8f){};

    // ---- A pipeline prologue: stage K-step 0 ----
    float4 ca0 = *(const float4*)(pA + 0);
    float4 ca1 = *(const float4*)(pA + 4);
    float4 ca2 = *(const float4*)(pA + 8);
    float4 ca3 = *(const float4*)(pA + 12);

#pragma unroll 1
    for (int ks = 0; ks < KSTEPS; ++ks) {
        // batch-load this K-step's 8 B fragments (immediate offsets nt*32768 B)
        v16bf barr[NT];
#pragma unroll
        for (int nt = 0; nt < NT; ++nt)
            barr[nt] = pBks[(size_t)nt * KSTEPS * 32];

        // prefetch next K-step's A (wraps harmlessly on the last iteration)
        const int kn = ((ks + 1) & (KSTEPS - 1)) * 32;
        const float4 na0 = *(const float4*)(pA + kn);
        const float4 na1 = *(const float4*)(pA + kn + 4);
        const float4 na2 = *(const float4*)(pA + kn + 8);
        const float4 na3 = *(const float4*)(pA + kn + 12);

        // convert staged A regs into this K-step's fragment
        v16bf afrag;
        afrag[0]  = (__bf16)ca0.x; afrag[1]  = (__bf16)ca0.y;
        afrag[2]  = (__bf16)ca0.z; afrag[3]  = (__bf16)ca0.w;
        afrag[4]  = (__bf16)ca1.x; afrag[5]  = (__bf16)ca1.y;
        afrag[6]  = (__bf16)ca1.z; afrag[7]  = (__bf16)ca1.w;
        afrag[8]  = (__bf16)ca2.x; afrag[9]  = (__bf16)ca2.y;
        afrag[10] = (__bf16)ca2.z; afrag[11] = (__bf16)ca2.w;
        afrag[12] = (__bf16)ca3.x; afrag[13] = (__bf16)ca3.y;
        afrag[14] = (__bf16)ca3.z; afrag[15] = (__bf16)ca3.w;
        ca0 = na0; ca1 = na1; ca2 = na2; ca3 = na3;

        // WMMA burst: loads complete in order -> descending partial waits
#pragma unroll
        for (int nt = 0; nt < NT; ++nt) {
            acc[nt] = __builtin_amdgcn_wmma_f32_16x16x32_bf16(
                false, afrag, false, barr[nt], (short)0, acc[nt], false, false);
        }

        pBks += 32;   // advance one K-step (32 lanes * 1 fragment)
    }

    // Epilogue: lane holds (M = r + 8*khalf, N = rc) of each 16x16 tile.
    float pos = 0.0f, neg = 0.0f;
#pragma unroll
    for (int nt = 0; nt < NT; ++nt) {
        const int kg = nt * 16 + rc;      // global ctx index
#pragma unroll
        for (int r = 0; r < 8; ++r) {
            const int s = s_base + r + 8 * khalf;
            if (s < len) {
                const float res = acc[nt][r];
                // softplus(-res), numerically stable, fast transcendentals
                const float sp = fmaxf(-res, 0.0f) + __logf(1.0f + __expf(-fabsf(res)));
                if (kg == b_idx) pos += LOG2F_ - sp;          // Ep
                else             neg += sp + res - LOG2F_;    // Eq
            }
        }
    }

    // wave32 reduction; lane 0 writes this wave's partial pair
    for (int off = 16; off > 0; off >>= 1) {
        pos += __shfl_down(pos, off, 32);
        neg += __shfl_down(neg, off, 32);
    }
    if (lane == 0) {
        partials[2 * mTile]     = pos;
        partials[2 * mTile + 1] = neg;
    }
}

// ---------------- Kernel 3: final reduction ----------------
__global__ __launch_bounds__(256) void finalize_kernel(const int* __restrict__ lens,
                                                       const float* __restrict__ partials,
                                                       int nPart, float* __restrict__ out) {
    __shared__ float sp[256];
    __shared__ float sn[256];
    __shared__ int   sl[256];
    const int t = threadIdx.x;
    float P = 0.0f, N = 0.0f;
    for (int i = t; i < nPart; i += 256) {
        P += partials[2 * i];
        N += partials[2 * i + 1];
    }
    sl[t] = (t < BD) ? lens[t] : 0;
    sp[t] = P; sn[t] = N;
    __syncthreads();
    for (int off = 128; off > 0; off >>= 1) {
        if (t < off) { sp[t] += sp[t + off]; sn[t] += sn[t + off]; sl[t] += sl[t + off]; }
        __syncthreads();
    }
    if (t == 0) {
        const float num_nodes = (float)sl[0];
        out[0] = sn[0] / (num_nodes * (float)(BD - 1)) - sp[0] / num_nodes;
    }
}

extern "C" void kernel_launch(void* const* d_in, const int* in_sizes, int n_in,
                              void* d_out, int out_size, void* d_ws, size_t ws_size,
                              hipStream_t stream) {
    const float* seq  = (const float*)d_in[0];  // [B,S,D] f32
    const float* ctx  = (const float*)d_in[1];  // [B,D]   f32
    const int*   mask = (const int*)d_in[2];    // [B,S]   i32
    float* out = (float*)d_out;

    // workspace layout
    int*    lens     = (int*)d_ws;                                  // 128 ints   @ 0
    __bf16* ctxb     = (__bf16*)((char*)d_ws + 512);                // 256 KB     @ 512
    float*  partials = (float*)((char*)d_ws + 512 + BD * DD * 2);   // 64 KB

    pack_ctx_kernel<<<NT * KSTEPS, 32, 0, stream>>>(ctx, ctxb);
    lens_kernel<<<BD, 256, 0, stream>>>(mask, lens);
    miloss_main<<<MTILES / 8, 256, 0, stream>>>(seq, ctxb, lens, partials);
    finalize_kernel<<<1, 256, 0, stream>>>(lens, partials, MTILES, out);
}
